// SlotAttention_5257039970512
// MI455X (gfx1250) — matrix operand order repaired and checked by reading, not compiled, not run on hardware
//
#include <hip/hip_runtime.h>
#include <math.h>

typedef __attribute__((ext_vector_type(16))) _Float16 v16h;
typedef __attribute__((ext_vector_type(8)))  _Float16 v8h;
typedef __attribute__((ext_vector_type(4)))  _Float16 v4h;
typedef __attribute__((ext_vector_type(8)))  float    v8f;

// ---------------------------------------------------------------------------
// conv0 3->64, 5x5 SAME, ReLU. Direct fp32 (tiny FLOPs), writes f16 [B][64][H][W].
// ---------------------------------------------------------------------------
__global__ __launch_bounds__(256) void conv0_kernel(
    const float* __restrict__ img, const float* __restrict__ w,
    const float* __restrict__ bias, _Float16* __restrict__ out) {
  __shared__ float wl[75];
  int gid = blockIdx.x * 256 + threadIdx.x;
  int x = gid & 63, y = (gid >> 6) & 63, co = (gid >> 12) & 63, b = gid >> 18;
  if (threadIdx.x < 75) wl[threadIdx.x] = w[co * 75 + threadIdx.x];  // co uniform per block
  __syncthreads();
  float acc = bias[co];
  for (int ci = 0; ci < 3; ++ci)
    for (int ky = 0; ky < 5; ++ky) {
      int iy = y + ky - 2;
      if (iy < 0 || iy >= 64) continue;
      for (int kx = 0; kx < 5; ++kx) {
        int ix = x + kx - 2;
        if (ix < 0 || ix >= 64) continue;
        acc += img[((b * 3 + ci) * 64 + iy) * 64 + ix] * wl[(ci * 5 + ky) * 5 + kx];
      }
    }
  out[gid] = (_Float16)fmaxf(acc, 0.f);
}

// ---------------------------------------------------------------------------
// One-shot weight pre-convert: fp32 [co][ci][5][5] -> f16 [tap][co][ci]
// so per-tap LDS staging in the conv is a contiguous b128 copy.
// ---------------------------------------------------------------------------
__global__ __launch_bounds__(256) void weight_prep_kernel(
    const float* __restrict__ w, _Float16* __restrict__ wf) {
  int gid = blockIdx.x * 256 + threadIdx.x;  // 25*4096 = 102400
  int off = gid >> 12, rem = gid & 4095;
  int co = rem >> 6, ci = rem & 63;
  int dy = off / 5, dx = off % 5;
  wf[gid] = (_Float16)w[((co * 64 + ci) * 5 + dy) * 5 + dx];
}

// ---------------------------------------------------------------------------
// 64->64 5x5 conv as implicit GEMM with v_wmma_f32_16x16x32_f16.
// WG = one image row (64 px) x 64 Cout; 8 waves = 4 mtiles x 2 pixel-pairs.
// K = 25 taps * 64 cin, 100 WMMAs/wave. All fragment slices are contiguous
// 16B-aligned LDS vectors (slab transposed to [x][cin], strides = 72 halfs).
// ---------------------------------------------------------------------------
__global__ __launch_bounds__(256) void conv_wmma_kernel(
    const _Float16* __restrict__ in, const _Float16* __restrict__ wf,
    const float* __restrict__ bias, _Float16* __restrict__ out) {
  __shared__ _Float16 slab[5][64][72];  // [tap row][x][cin], 16B-aligned slices
  __shared__ _Float16 wt[64][72];       // [cout][cin] for current tap
  int tid = threadIdx.x;
  int b = blockIdx.x >> 6;
  int y = blockIdx.x & 63;

  // Stage 5-row input slab, transposed to [x][cin]; zero out-of-bounds rows.
  for (int r = 0; r < 5; ++r) {
    int gy = y + r - 2;
    bool ok = (gy >= 0) && (gy < 64);
    int gyc = ok ? gy : 0;
    for (int e = tid; e < 1024; e += 256) {
      int x = e & 63, c4 = (e >> 6) * 4;
      v4h val = {};
      if (ok) {
        const _Float16* src = in + (((b * 64 + c4) * 64 + gyc) * 64 + x);
        val[0] = src[0];
        val[1] = src[4096];
        val[2] = src[8192];
        val[3] = src[12288];
      }
      *(v4h*)&slab[r][x][c4] = val;
    }
  }

  int wid = tid >> 5, lane = tid & 31;
  int mtile = wid & 3, npair = wid >> 2;
  int lh = lane >> 4, ln16 = lane & 15;
  v8f c0 = {}, c1 = {};
  const v16h zero16 = {};

  for (int off = 0; off < 25; ++off) {
    int dy = off / 5, dx = off % 5;
    __syncthreads();  // prior tap's wt reads done (covers slab staging at off==0)
    // contiguous b128 copy of this tap's 64x64 f16 weights
    for (int e = tid; e < 512; e += 256) {
      int co = e >> 3, k8 = (e & 7) * 8;
      *(v8h*)&wt[co][k8] = *(const v8h*)&wf[off * 4096 + co * 64 + k8];
    }
    __syncthreads();

    int xb0 = 32 * npair + ln16 + dx - 2;
    int xb1 = xb0 + 16;
    int x0c = min(max(xb0, 0), 63), x1c = min(max(xb1, 0), 63);
    bool ok0 = (xb0 >= 0) && (xb0 < 64), ok1 = (xb1 >= 0) && (xb1 < 64);
    int am = 16 * mtile + ln16;
#pragma unroll
    for (int kb = 0; kb < 2; ++kb) {
      int kbase = kb * 32;
      // A (16x32 f16): lane<16 -> K 0..7 / 16..23 ; lane>=16 -> K 8..15 / 24..31
      v8h alo = *(const v8h*)&wt[am][kbase + lh * 8];
      v8h ahi = *(const v8h*)&wt[am][kbase + 16 + lh * 8];
      // B (32x16 f16): lane<16 -> K 0..7 / 16..23 ; lane>=16 -> K 8..15 / 24..31 ; N = lane&15
      const v8h* b0p = (const v8h*)&slab[dy][x0c][kbase + lh * 8];
      const v8h* b1p = (const v8h*)&slab[dy][x1c][kbase + lh * 8];
      v8h bl0 = b0p[0], bh0 = b0p[2];
      v8h bl1 = b1p[0], bh1 = b1p[2];
      v16h a, b0, b1;
#pragma unroll
      for (int e = 0; e < 8; ++e) {
        a[e] = alo[e];  a[8 + e] = ahi[e];
        b0[e] = bl0[e]; b0[8 + e] = bh0[e];
        b1[e] = bl1[e]; b1[8 + e] = bh1[e];
      }
      b0 = ok0 ? b0 : zero16;
      b1 = ok1 ? b1 : zero16;
      c0 = __builtin_amdgcn_wmma_f32_16x16x32_f16(false, a, false, b0, (short)0, c0, false, false);
      c1 = __builtin_amdgcn_wmma_f32_16x16x32_f16(false, a, false, b1, (short)0, c1, false, false);
    }
  }

  // C 16x16 layout: VGPR r holds M=r (lanes 0-15) / M=r+8 (lanes 16-31)
#pragma unroll
  for (int r = 0; r < 8; ++r) {
    int co = 16 * mtile + r + 8 * lh;
    float bv = bias[co];
    float v0 = fmaxf(c0[r] + bv, 0.f);
    float v1 = fmaxf(c1[r] + bv, 0.f);
    int base = ((b * 64 + co) * 64 + y) * 64;
    out[base + 32 * npair + ln16] = (_Float16)v0;
    out[base + 32 * npair + 16 + ln16] = (_Float16)v1;
  }
}

// ---------------------------------------------------------------------------
// Generic WMMA GEMM: out[R x 64] = act( X[R x 64] * W[64 x 64]^T + bias )
// WG: 128 rows, 8 waves, each wave 16 rows x 64 cols (8 wmma, K=64).
// ---------------------------------------------------------------------------
__global__ __launch_bounds__(256) void gemm64_kernel(
    const _Float16* __restrict__ X, const float* __restrict__ W,
    const float* __restrict__ bias, _Float16* __restrict__ out, int relu) {
  __shared__ _Float16 xt[128][72];
  __shared__ _Float16 wt[64][72];
  int tid = threadIdx.x;
  int rowbase = blockIdx.x * 128;
  for (int e = tid; e < 2048; e += 256) {   // v4h copies
    int r = e >> 4, k4 = (e & 15) * 4;
    *(v4h*)&xt[r][k4] = *(const v4h*)&X[(rowbase + r) * 64 + k4];
  }
  for (int e = tid; e < 1024; e += 256) {   // fp32 -> f16 convert, vectorized
    int n = e >> 4, k4 = (e & 15) * 4;
    float4 f = *(const float4*)&W[n * 64 + k4];
    v4h h; h[0] = (_Float16)f.x; h[1] = (_Float16)f.y; h[2] = (_Float16)f.z; h[3] = (_Float16)f.w;
    *(v4h*)&wt[n][k4] = h;
  }
  __syncthreads();

  int wid = tid >> 5, lane = tid & 31, lh = lane >> 4, ln16 = lane & 15;
  int arow = 16 * wid + ln16;
  v8f c[4] = {};
#pragma unroll
  for (int kb = 0; kb < 2; ++kb) {
    int kbase = kb * 32;
    v8h alo = *(const v8h*)&xt[arow][kbase + lh * 8];
    v8h ahi = *(const v8h*)&xt[arow][kbase + 16 + lh * 8];
    v16h a;
#pragma unroll
    for (int e = 0; e < 8; ++e) { a[e] = alo[e]; a[8 + e] = ahi[e]; }
#pragma unroll
    for (int nt = 0; nt < 4; ++nt) {
      int n = 16 * nt + ln16;
      const v8h* bp = (const v8h*)&wt[n][kbase + lh * 8];
      v8h bl = bp[0], bh = bp[2];
      v16h bf;
#pragma unroll
      for (int e = 0; e < 8; ++e) { bf[e] = bl[e]; bf[8 + e] = bh[e]; }
      c[nt] = __builtin_amdgcn_wmma_f32_16x16x32_f16(false, a, false, bf, (short)0, c[nt], false, false);
    }
  }
#pragma unroll
  for (int nt = 0; nt < 4; ++nt) {
#pragma unroll
    for (int r = 0; r < 8; ++r) {
      int grow = rowbase + 16 * wid + r + 8 * lh;
      int col = 16 * nt + ln16;
      float v = c[nt][r] + (bias ? bias[col] : 0.f);
      if (relu) v = fmaxf(v, 0.f);
      out[grow * 64 + col] = (_Float16)v;
    }
  }
}

// ---------------------------------------------------------------------------
// Pos-embed + joint (N,D) LayerNorm stats (per image), then normalize.
// ---------------------------------------------------------------------------
__global__ __launch_bounds__(256) void stats_kernel(
    const _Float16* __restrict__ act, const float* __restrict__ pw,
    const float* __restrict__ pb, float* __restrict__ stats) {
  __shared__ float spw[256], spb[64];
  __shared__ float rs[256], rq[256];
  int b = blockIdx.x, tid = threadIdx.x;
  spw[tid] = pw[tid];
  if (tid < 64) spb[tid] = pb[tid];
  __syncthreads();
  float s = 0.f, q = 0.f;
  for (int e = tid; e < 262144; e += 256) {
    int d = e >> 12, yy = (e >> 6) & 63, xx = e & 63;
    float ry = yy * (1.f / 63.f), cx = xx * (1.f / 63.f);
    float pos = spw[d * 4] * ry + spw[d * 4 + 1] * cx +
                spw[d * 4 + 2] * (1.f - ry) + spw[d * 4 + 3] * (1.f - cx) + spb[d];
    float v = (float)act[((b * 64 + d) * 64 + yy) * 64 + xx] + pos;
    s += v; q += v * v;
  }
  rs[tid] = s; rq[tid] = q;
  __syncthreads();
  for (int st = 128; st > 0; st >>= 1) {
    if (tid < st) { rs[tid] += rs[tid + st]; rq[tid] += rq[tid + st]; }
    __syncthreads();
  }
  if (tid == 0) {
    float mean = rs[0] * (1.f / 262144.f);
    float var = rq[0] * (1.f / 262144.f) - mean * mean;
    stats[b * 2] = mean;
    stats[b * 2 + 1] = rsqrtf(var + 1e-5f);
  }
}

__global__ __launch_bounds__(256) void posnorm_kernel(
    const _Float16* __restrict__ act, const float* __restrict__ pw,
    const float* __restrict__ pb, const float* __restrict__ stats,
    _Float16* __restrict__ xn) {
  int gid = blockIdx.x * 256 + threadIdx.x;
  int d = gid & 63, n = (gid >> 6) & 4095, b = gid >> 18;
  int yy = n >> 6, xx = n & 63;
  float ry = yy * (1.f / 63.f), cx = xx * (1.f / 63.f);
  float pos = pw[d * 4] * ry + pw[d * 4 + 1] * cx +
              pw[d * 4 + 2] * (1.f - ry) + pw[d * 4 + 3] * (1.f - cx) + pb[d];
  float v = (float)act[((b * 64 + d) * 64 + yy) * 64 + xx] + pos;
  xn[gid] = (_Float16)((v - stats[b * 2]) * stats[b * 2 + 1]);  // [b][n][d]
}

// Per-token LayerNorm (eps 1e-3, affine) over D=64
__global__ __launch_bounds__(256) void ln_token_kernel(
    const _Float16* __restrict__ X, const float* __restrict__ gw,
    const float* __restrict__ gb, _Float16* __restrict__ out) {
  int t = blockIdx.x * 256 + threadIdx.x;
  const v8h* xp = (const v8h*)(X + (size_t)t * 64);
  float v[64]; float s = 0.f, q = 0.f;
#pragma unroll
  for (int i = 0; i < 8; ++i) {
    v8h c8 = xp[i];
#pragma unroll
    for (int j = 0; j < 8; ++j) { float f = (float)c8[j]; v[i * 8 + j] = f; s += f; q += f * f; }
  }
  float mu = s * (1.f / 64.f);
  float rstd = rsqrtf(q * (1.f / 64.f) - mu * mu + 1e-3f);
  v8h* op = (v8h*)(out + (size_t)t * 64);
#pragma unroll
  for (int i = 0; i < 8; ++i) {
    v8h o;
#pragma unroll
    for (int j = 0; j < 8; ++j)
      o[j] = (_Float16)((v[i * 8 + j] - mu) * rstd * gw[i * 8 + j] + gb[i * 8 + j]);
    op[i] = o;
  }
}

// ---------------------------------------------------------------------------
// Slot attention loop (small, VALU).
// ---------------------------------------------------------------------------
__global__ void init_slots_kernel(const float* __restrict__ noise, const float* __restrict__ mu,
                                  const float* __restrict__ lsig, float* __restrict__ slots) {
  int gid = blockIdx.x * 256 + threadIdx.x;  // 28672
  int d = gid & 63;
  slots[gid] = mu[d] + expf(lsig[d]) * noise[gid];
}

__global__ void slot_q_kernel(const float* __restrict__ slots, const float* __restrict__ lnw,
                              const float* __restrict__ lnb, const float* __restrict__ qw,
                              float* __restrict__ q) {
  __shared__ float sh[64];
  __shared__ float s_mu, s_rs;
  int bs = blockIdx.x, t = threadIdx.x;
  float v = slots[bs * 64 + t];
  sh[t] = v;
  __syncthreads();
  if (t == 0) {
    float s = 0.f, qq = 0.f;
    for (int k = 0; k < 64; ++k) { s += sh[k]; qq += sh[k] * sh[k]; }
    float mu = s / 64.f;
    s_mu = mu; s_rs = rsqrtf(qq / 64.f - mu * mu + 1e-3f);
  }
  __syncthreads();
  float lnv = (v - s_mu) * s_rs * lnw[t] + lnb[t];
  sh[t] = lnv;
  __syncthreads();
  float acc = 0.f;
  for (int k = 0; k < 64; ++k) acc += sh[k] * qw[t * 64 + k];
  q[bs * 64 + t] = acc * 0.125f;  // fold scale = D^-0.5
}

__global__ __launch_bounds__(256) void dots_attn_kernel(
    const float* __restrict__ q, const _Float16* __restrict__ kbuf,
    float* __restrict__ attn) {
  __shared__ float qs[448];
  int b = blockIdx.x, t = threadIdx.x;
  int n = blockIdx.y * 256 + t;
  for (int e = t; e < 448; e += 256) qs[e] = q[b * 448 + e];
  __syncthreads();
  const v8h* kp = (const v8h*)(kbuf + (size_t)(b * 4096 + n) * 64);
  float kr[64];
#pragma unroll
  for (int i = 0; i < 8; ++i) {
    v8h c8 = kp[i];
#pragma unroll
    for (int j = 0; j < 8; ++j) kr[i * 8 + j] = (float)c8[j];
  }
  float ds[7]; float m = -1e30f;
  for (int s = 0; s < 7; ++s) {
    float acc = 0.f;
    for (int k = 0; k < 64; ++k) acc += qs[s * 64 + k] * kr[k];
    ds[s] = acc; m = fmaxf(m, acc);
  }
  float tot = 0.f;
  for (int s = 0; s < 7; ++s) { ds[s] = expf(ds[s] - m); tot += ds[s]; }
  float inv = 1.f / tot;
  for (int s = 0; s < 7; ++s) attn[(b * 7 + s) * 4096 + n] = ds[s] * inv + 1e-8f;
}

__global__ __launch_bounds__(256) void rowsum_kernel(const float* __restrict__ attn,
                                                     float* __restrict__ rowsum) {
  __shared__ float red[256];
  int bs = blockIdx.x, t = threadIdx.x;
  float s = 0.f;
  for (int j = t; j < 4096; j += 256) s += attn[bs * 4096 + j];
  red[t] = s;
  __syncthreads();
  for (int st = 128; st > 0; st >>= 1) {
    if (t < st) red[t] += red[t + st];
    __syncthreads();
  }
  if (t == 0) rowsum[bs] = red[0];
}

__global__ __launch_bounds__(256) void updates_kernel(
    const float* __restrict__ attn, const _Float16* __restrict__ vbuf,
    const float* __restrict__ rowsum, float* __restrict__ upd) {
  __shared__ float sh[256];
  int bs = blockIdx.x, t = threadIdx.x;
  int b = bs / 7;
  int d = t & 63, ch = t >> 6;
  float acc = 0.f;
  for (int j = ch * 1024; j < ch * 1024 + 1024; ++j)
    acc += attn[bs * 4096 + j] * (float)vbuf[(b * 4096 + j) * 64 + d];
  sh[t] = acc;
  __syncthreads();
  if (ch == 0) upd[bs * 64 + d] = (sh[d] + sh[64 + d] + sh[128 + d] + sh[192 + d]) / rowsum[bs];
}

__global__ void gru_mlp_kernel(
    const float* __restrict__ upd_in, const float* __restrict__ src,
    const float* __restrict__ wih, const float* __restrict__ whh,
    const float* __restrict__ bih, const float* __restrict__ bhh,
    const float* __restrict__ lnw, const float* __restrict__ lnb,
    const float* __restrict__ m1w, const float* __restrict__ m1b,
    const float* __restrict__ m2w, const float* __restrict__ m2b,
    float* __restrict__ dst) {
  __shared__ float su[64], sp[64], gi[192], gh[192], sn[64], hl[64], m1[128];
  __shared__ float s_mu, s_rs;
  int bs = blockIdx.x, t = threadIdx.x;  // 192 threads
  if (t < 64) { su[t] = upd_in[bs * 64 + t]; sp[t] = src[bs * 64 + t]; }
  __syncthreads();
  {
    float a = bih[t], c = bhh[t];
    for (int k = 0; k < 64; ++k) { a += su[k] * wih[t * 64 + k]; c += sp[k] * whh[t * 64 + k]; }
    gi[t] = a; gh[t] = c;
  }
  __syncthreads();
  if (t < 64) {
    float r = 1.f / (1.f + expf(-(gi[t] + gh[t])));
    float z = 1.f / (1.f + expf(-(gi[64 + t] + gh[64 + t])));
    float nn = tanhf(gi[128 + t] + r * gh[128 + t]);
    sn[t] = (1.f - z) * nn + z * sp[t];
  }
  __syncthreads();
  if (t == 0) {
    float s = 0.f, q = 0.f;
    for (int k = 0; k < 64; ++k) { s += sn[k]; q += sn[k] * sn[k]; }
    float mu = s / 64.f;
    s_mu = mu; s_rs = rsqrtf(q / 64.f - mu * mu + 1e-3f);
  }
  __syncthreads();
  if (t < 64) hl[t] = (sn[t] - s_mu) * s_rs * lnw[t] + lnb[t];
  __syncthreads();
  if (t < 128) {
    float acc = m1b[t];
    for (int k = 0; k < 64; ++k) acc += hl[k] * m1w[t * 64 + k];
    m1[t] = fmaxf(acc, 0.f);
  }
  __syncthreads();
  if (t < 64) {
    float acc = m2b[t];
    for (int k = 0; k < 128; ++k) acc += m1[k] * m2w[t * 128 + k];
    dst[bs * 64 + t] = sn[t] + acc;
  }
}

// ---------------------------------------------------------------------------
extern "C" void kernel_launch(void* const* d_in, const int* in_sizes, int n_in,
                              void* d_out, int out_size, void* d_ws, size_t ws_size,
                              hipStream_t stream) {
  (void)in_sizes; (void)n_in; (void)out_size; (void)ws_size;
  const float* image = (const float*)d_in[0];
  const float* noise = (const float*)d_in[1];
  const float* c0w = (const float*)d_in[2];
  const float* c0b = (const float*)d_in[3];
  const float* c1w = (const float*)d_in[4];
  const float* c1b = (const float*)d_in[5];
  const float* c2w = (const float*)d_in[6];
  const float* c2b = (const float*)d_in[7];
  const float* c3w = (const float*)d_in[8];
  const float* c3b = (const float*)d_in[9];
  const float* posw = (const float*)d_in[10];
  const float* posb = (const float*)d_in[11];
  const float* fc1w = (const float*)d_in[12];
  const float* fc1b = (const float*)d_in[13];
  const float* fc2w = (const float*)d_in[14];
  const float* fc2b = (const float*)d_in[15];
  const float* smu = (const float*)d_in[16];
  const float* slsig = (const float*)d_in[17];
  const float* qw = (const float*)d_in[18];
  const float* kw = (const float*)d_in[19];
  const float* vw = (const float*)d_in[20];
  const float* wih = (const float*)d_in[21];
  const float* whh = (const float*)d_in[22];
  const float* bih = (const float*)d_in[23];
  const float* bhh = (const float*)d_in[24];
  const float* m1w = (const float*)d_in[25];
  const float* m1b = (const float*)d_in[26];
  const float* m2w = (const float*)d_in[27];
  const float* m2b = (const float*)d_in[28];
  const float* lninw = (const float*)d_in[29];
  const float* lninb = (const float*)d_in[30];
  const float* lnsw = (const float*)d_in[31];
  const float* lnsb = (const float*)d_in[32];
  const float* lnfw = (const float*)d_in[33];
  const float* lnfb = (const float*)d_in[34];

  // Workspace layout (~108.6 MB total)
  _Float16* bufA  = (_Float16*)d_ws;          // 16,777,216 f16
  _Float16* bufB  = bufA + 16777216;
  _Float16* bufC  = bufB + 16777216;
  _Float16* wprep = bufC + 16777216;          // 102,400 f16 (25 taps x 64x64)
  float* attn   = (float*)(wprep + 102400);   // [B,7,4096]
  float* qbuf   = attn + 1835008;             // [B,7,64]
  float* rowsum = qbuf + 28672;               // [B*7]
  float* updbuf = rowsum + 448;               // [B,7,64]
  float* slots  = updbuf + 28672;             // [B,7,64]
  float* stats  = slots + 28672;              // [B,2]

  // Encoder
  conv0_kernel<<<65536, 256, 0, stream>>>(image, c0w, c0b, bufA);
  weight_prep_kernel<<<400, 256, 0, stream>>>(c1w, wprep);
  conv_wmma_kernel<<<4096, 256, 0, stream>>>(bufA, wprep, c1b, bufB);
  weight_prep_kernel<<<400, 256, 0, stream>>>(c2w, wprep);
  conv_wmma_kernel<<<4096, 256, 0, stream>>>(bufB, wprep, c2b, bufA);
  weight_prep_kernel<<<400, 256, 0, stream>>>(c3w, wprep);
  conv_wmma_kernel<<<4096, 256, 0, stream>>>(bufA, wprep, c3b, bufB);
  stats_kernel<<<64, 256, 0, stream>>>(bufB, posw, posb, stats);
  posnorm_kernel<<<65536, 256, 0, stream>>>(bufB, posw, posb, stats, bufA);
  gemm64_kernel<<<2048, 256, 0, stream>>>(bufA, fc1w, fc1b, bufB, 1);
  gemm64_kernel<<<2048, 256, 0, stream>>>(bufB, fc2w, fc2b, bufC, 0);
  ln_token_kernel<<<1024, 256, 0, stream>>>(bufC, lninw, lninb, bufA);
  gemm64_kernel<<<2048, 256, 0, stream>>>(bufA, kw, nullptr, bufB, 0);  // k
  gemm64_kernel<<<2048, 256, 0, stream>>>(bufA, vw, nullptr, bufC, 0);  // v

  // Slot attention
  init_slots_kernel<<<112, 256, 0, stream>>>(noise, smu, slsig, slots);
  for (int it = 0; it < 3; ++it) {
    slot_q_kernel<<<448, 64, 0, stream>>>(slots, lnsw, lnsb, qw, qbuf);
    dots_attn_kernel<<<dim3(64, 16), 256, 0, stream>>>(qbuf, bufB, attn);
    rowsum_kernel<<<448, 256, 0, stream>>>(attn, rowsum);
    updates_kernel<<<448, 256, 0, stream>>>(attn, bufC, rowsum, updbuf);
    float* dst = (it == 2) ? (float*)d_out : slots;
    gru_mlp_kernel<<<448, 192, 0, stream>>>(updbuf, slots, wih, whh, bih, bhh,
                                            lnfw, lnfb, m1w, m1b, m2w, m2b, dst);
  }
}
